// PN2_43997644980871
// MI455X (gfx1250) — matrix operand order repaired and checked by reading, not compiled, not run on hardware
//
#include <hip/hip_runtime.h>
#include <hip/hip_bf16.h>
#include <stdint.h>

typedef _Float16 half_t;
typedef __attribute__((ext_vector_type(16))) _Float16 v16h;
typedef __attribute__((ext_vector_type(8)))  float    v8f;

#define BATCH 32
#define N0    1000
#define S1    256
#define K1    32
#define S2    128
#define K2    64
#define S3    50
#define K3    64

// ---------------------------------------------------------------------------
// Fold BN (eval) into f16 weights/bias:  y = (s*W)x + (s*(b-mean)+beta)
// Pads Cout->cpad and Cin->kpad with zeros so GEMMs run on clean multiples.
// ---------------------------------------------------------------------------
__global__ void prep_weights_kernel(const float* __restrict__ w,
                                    const float* __restrict__ b,
                                    const float* __restrict__ gamma,
                                    const float* __restrict__ beta,
                                    const float* __restrict__ mean,
                                    const float* __restrict__ var,
                                    half_t* __restrict__ wf,
                                    float* __restrict__ bf,
                                    int cout, int cin, int cpad, int kpad)
{
    int t = blockIdx.x * blockDim.x + threadIdx.x;
    if (t >= cpad * kpad) return;
    int o = t / kpad;
    int k = t - o * kpad;
    float s = 1.f, bb = 0.f;
    if (o < cout) {
        if (gamma) {
            s  = gamma[o] * rsqrtf(var[o] + 1e-5f);
            bb = (b[o] - mean[o]) * s + beta[o];
        } else {
            bb = b[o];
        }
    }
    float val = (o < cout && k < cin) ? w[(size_t)o * cin + k] * s : 0.f;
    wf[(size_t)o * kpad + k] = (half_t)val;
    if (k == 0) bf[o] = bb;
}

// ---------------------------------------------------------------------------
// Iterative farthest-point sampling, one block per batch (serial in S,
// parallel block argmax over N). Writes sampled coordinates directly.
// ---------------------------------------------------------------------------
__global__ void fps_kernel(const float* __restrict__ xyz, int N, int S,
                           float* __restrict__ new_xyz)
{
    __shared__ float dist[1024];
    __shared__ float sval[256];
    __shared__ int   sidx[256];
    __shared__ int   scur;
    int b = blockIdx.x, tid = threadIdx.x;
    const float* px = xyz + (size_t)b * N * 3;
    for (int i = tid; i < N; i += 256) dist[i] = 1e10f;
    if (tid == 0) scur = 0;
    __syncthreads();
    for (int s = 0; s < S; ++s) {
        int cur = scur;
        float cx = px[cur * 3 + 0], cy = px[cur * 3 + 1], cz = px[cur * 3 + 2];
        if (tid == 0) {
            float* o = new_xyz + ((size_t)b * S + s) * 3;
            o[0] = cx; o[1] = cy; o[2] = cz;
        }
        float best = -1.f; int bi = 0;
        for (int i = tid; i < N; i += 256) {
            float dx = px[i*3+0]-cx, dy = px[i*3+1]-cy, dz = px[i*3+2]-cz;
            float nd = fminf(dist[i], dx*dx + dy*dy + dz*dz);
            dist[i] = nd;
            if (nd > best) { best = nd; bi = i; }
        }
        sval[tid] = best; sidx[tid] = bi;
        __syncthreads();
        for (int off = 128; off > 0; off >>= 1) {
            if (tid < off && sval[tid + off] > sval[tid]) {
                sval[tid] = sval[tid + off]; sidx[tid] = sidx[tid + off];
            }
            __syncthreads();
        }
        if (tid == 0) scur = sidx[0];
        __syncthreads();
    }
}

// ---------------------------------------------------------------------------
// Ball query: first K in-radius neighbors in ascending index order,
// padded with the first hit (matches reference sort/pad semantics).
// ---------------------------------------------------------------------------
__global__ void ball_query_kernel(const float* __restrict__ xyz,
                                  const float* __restrict__ new_xyz,
                                  int N, int S, int K, float r2,
                                  int* __restrict__ idx)
{
    int t = blockIdx.x * blockDim.x + threadIdx.x;
    if (t >= BATCH * S) return;
    int b = t / S;
    const float* px = xyz + (size_t)b * N * 3;
    const float* c  = new_xyz + (size_t)t * 3;
    float cx = c[0], cy = c[1], cz = c[2];
    int* out = idx + (size_t)t * K;
    int cnt = 0, first = 0; bool has = false;
    for (int j = 0; j < N; ++j) {
        float dx = px[j*3]-cx, dy = px[j*3+1]-cy, dz = px[j*3+2]-cz;
        if (dx*dx + dy*dy + dz*dz <= r2) {
            if (!has) { first = j; has = true; }
            out[cnt++] = j;
            if (cnt >= K) break;
        }
    }
    for (int q = cnt; q < K; ++q) out[q] = first;
}

// ---------------------------------------------------------------------------
// Build grouped activation matrix X[M, kpad] (f16): cols 0..2 = centered xyz,
// cols 3..3+Cf-1 = gathered features, rest zero-padded for the WMMA K dim.
// ---------------------------------------------------------------------------
__global__ void group_kernel(const float* __restrict__ xyz,
                             const float* __restrict__ new_xyz,
                             const half_t* __restrict__ feats,
                             const int* __restrict__ idx,
                             int N, int S, int K, int Cf, int kpad,
                             half_t* __restrict__ X)
{
    size_t t = (size_t)blockIdx.x * blockDim.x + threadIdx.x;
    size_t total = (size_t)BATCH * S * K * kpad;
    if (t >= total) return;
    int    c  = (int)(t % kpad);
    size_t m  = t / kpad;
    int    k  = (int)(m % K);
    size_t bs = m / K;                 // b*S + s
    int    b  = (int)(bs / S);
    int    j  = idx[bs * K + k];
    float val = 0.f;
    if (c < 3)            val = xyz[((size_t)b * N + j) * 3 + c] - new_xyz[bs * 3 + c];
    else if (c < 3 + Cf)  val = (float)feats[((size_t)b * N + j) * Cf + (c - 3)];
    X[m * kpad + c] = (half_t)val;
}

// ---------------------------------------------------------------------------
// WMMA GEMM: D = A[M,Ka] * W^T[Ka,Npad] + bias, optional ReLU.
// Each wave owns a 32x64 strip (2 M-tiles x 4 N-tiles = 8 accumulators).
// Per 32-deep k-step: 2 A-fragment loads + 4 B-fragment loads feed
// 8 v_wmma_f32_16x16x32_f16, maximizing fragment reuse (A x4, B x2) so the
// GEMMs stay under the 23.3 TB/s HBM roof instead of re-streaming A per tile.
// Per-lane fragment addressing follows the CDNA5 ISA VGPR layout tables.
// ---------------------------------------------------------------------------
union AFrag { v16h h; uint32_t u[8]; };

__global__ void __launch_bounds__(256)
wmma_gemm_kernel(const half_t* __restrict__ A,
                 const half_t* __restrict__ W,
                 const float*  __restrict__ bias,
                 half_t* __restrict__ outH,
                 float*  __restrict__ outF,
                 int M, int Ka, int Npad, int Nreal, int relu)
{
    const int groups_n = Npad >> 6;                // N strips of 64 (all cpads % 64 == 0)
    const int tiles    = (M >> 5) * groups_n;      // 32 M-rows per wave
    int wid = (blockIdx.x * blockDim.x + threadIdx.x) >> 5;   // wave-uniform
    if (wid >= tiles) return;                                  // whole-wave exit
    int lane = threadIdx.x & 31;
    int hi   = lane >> 4;
    int l15  = lane & 15;
    int tm = wid / groups_n;
    int tg = wid - tm * groups_n;
    int m0   = tm << 5;
    int col0 = (tg << 6) + l15;                    // column of N sub-tile 0

    const half_t* arow0 = A + (size_t)(m0 + l15) * Ka;        // rows m0..m0+15
    const half_t* arow1 = A + (size_t)(m0 + 16 + l15) * Ka;   // rows m0+16..m0+31
    const half_t* wrow0 = W + (size_t)col0 * Ka;              // B col n -> W row n

    v8f acc[2][4];
#pragma unroll
    for (int mt = 0; mt < 2; ++mt)
#pragma unroll
        for (int nt = 0; nt < 4; ++nt) acc[mt][nt] = (v8f){};

    for (int kk = 0; kk < Ka; kk += 32) {
        if (kk + 32 < Ka) {                        // gfx1250 global_prefetch_b8
            __builtin_prefetch((const void*)(arow0 + kk + 32), 0, 1);
            __builtin_prefetch((const void*)(arow1 + kk + 32), 0, 1);
        }
        AFrag a0, a1;
#pragma unroll
        for (int v = 0; v < 8; ++v) {
            int ka = kk + 2 * v + (hi << 3) + ((v >= 4) ? 8 : 0); // A 16x32 layout
            a0.u[v] = *reinterpret_cast<const uint32_t*>(arow0 + ka);
            a1.u[v] = *reinterpret_cast<const uint32_t*>(arow1 + ka);
        }
#pragma unroll
        for (int nt = 0; nt < 4; ++nt) {
            AFrag bf;
            const half_t* wrow = wrow0 + (size_t)(nt * 16) * Ka;
#pragma unroll
            for (int v = 0; v < 8; ++v) {
                int kb = kk + 2 * v + (hi << 3) + ((v >= 4) ? 8 : 0); // B 32x16 layout
                bf.u[v] = *reinterpret_cast<const uint32_t*>(wrow + kb);
            }
            acc[0][nt] = __builtin_amdgcn_wmma_f32_16x16x32_f16(false, a0.h, false, bf.h,
                                                                (short)0, acc[0][nt], false, false);
            acc[1][nt] = __builtin_amdgcn_wmma_f32_16x16x32_f16(false, a1.h, false, bf.h,
                                                                (short)0, acc[1][nt], false, false);
        }
    }
#pragma unroll
    for (int nt = 0; nt < 4; ++nt) {
        int col = col0 + nt * 16;
        float bv = bias ? bias[col] : 0.f;
#pragma unroll
        for (int mt = 0; mt < 2; ++mt) {
#pragma unroll
            for (int r = 0; r < 8; ++r) {
                int mm = m0 + mt * 16 + r + (hi << 3);   // D: m = r + 8*hi, n = lane&15
                float v = acc[mt][nt][r] + bv;
                if (relu) v = fmaxf(v, 0.f);
                if (outH) outH[(size_t)mm * Npad + col] = (half_t)v;
                if (outF && col < Nreal) outF[(size_t)mm * Nreal + col] = v;
            }
        }
    }
}

// ---------------------------------------------------------------------------
// Max-pool over the K (nsample) axis: [B*S, K, C(stride)] -> [B*S, C] f16.
// ---------------------------------------------------------------------------
__global__ void maxpool_kernel(const half_t* __restrict__ h, int S, int K,
                               int C, int stride, half_t* __restrict__ out)
{
    int t = blockIdx.x * blockDim.x + threadIdx.x;
    if (t >= BATCH * S * C) return;
    int c  = t % C;
    int bs = t / C;
    const half_t* p = h + (size_t)bs * K * stride + c;
    float m = -1e30f;
    for (int k = 0; k < K; ++k) m = fmaxf(m, (float)p[(size_t)k * stride]);
    out[(size_t)bs * C + c] = (half_t)m;
}

// ---------------------------------------------------------------------------
extern "C" void kernel_launch(void* const* d_in, const int* in_sizes, int n_in,
                              void* d_out, int out_size, void* d_ws, size_t ws_size,
                              hipStream_t stream)
{
    (void)out_size;
    if (n_in < 63) return;

    // part_pcs is the only leaf with 32*1000*3 = 96000 elements.
    const float* part_pcs = (const float*)d_in[n_in - 1];
    for (int i = 0; i < n_in; ++i)
        if (in_sizes[i] == BATCH * N0 * 3) part_pcs = (const float*)d_in[i];

    // ---- workspace carve ----
    uint8_t* ws = (uint8_t*)d_ws;
    size_t off = 0;
    auto take = [&](size_t bytes) -> void* {
        void* p = ws + off;
        off = (off + bytes + 255) & ~(size_t)255;
        return p;
    };
    half_t* bufA   = (half_t*)take((size_t)262144 * 160 * 2); // largest A-side tensor
    half_t* bufB   = (half_t*)take((size_t)262144 * 256 * 2); // largest B-side tensor
    int*    idx1   = (int*)   take((size_t)BATCH * S1 * K1 * 4);
    int*    idx2   = (int*)   take((size_t)BATCH * S2 * K2 * 4);
    int*    idx3   = (int*)   take((size_t)BATCH * S3 * K3 * 4);
    float*  l1_xyz = (float*) take((size_t)BATCH * S1 * 3 * 4);
    float*  l2_xyz = (float*) take((size_t)BATCH * S2 * 3 * 4);
    half_t* l1_pts = (half_t*)take((size_t)BATCH * S1 * 128 * 2);
    half_t* l2_pts = (half_t*)take((size_t)BATCH * S2 * 256 * 2);
    half_t* l3_pts = (half_t*)take((size_t)BATCH * S3 * 512 * 2);
    half_t* gf16   = (half_t*)take((size_t)BATCH * S3 * 64 * 2);
    half_t* fc1o   = (half_t*)take((size_t)1600 * 256 * 2);
    half_t* fc2o   = (half_t*)take((size_t)1600 * 512 * 2);
    half_t* wreg   = (half_t*)take((size_t)600000 * 2);
    float*  breg   = (float*) take((size_t)4096 * 4);
    if (off > ws_size) return;

    // ---- param leaf indices (jax pytree flatten: dict keys sorted) ----
    // 0..7: conv6{b,w} fc1{b,w} fc2{b,w} fc3{b,w}
    // 8..61: sa1/sa2/sa3, 3 layers each of {b,beta,gamma,mean,var,w}
    struct LW { int wi, bi, gi, bei, mi, vi, cout, cin, cpad, kpad; };
    const LW lws[13] = {
        {13,  8, 10,  9, 11, 12,  64,   3,  64,  32},   // sa1.0
        {19, 14, 16, 15, 17, 18,  64,  64,  64,  64},   // sa1.1
        {25, 20, 22, 21, 23, 24, 128,  64, 128,  64},   // sa1.2
        {31, 26, 28, 27, 29, 30, 128, 131, 128, 160},   // sa2.0
        {37, 32, 34, 33, 35, 36, 128, 128, 128, 128},   // sa2.1
        {43, 38, 40, 39, 41, 42, 256, 128, 256, 128},   // sa2.2
        {49, 44, 46, 45, 47, 48, 256, 259, 256, 288},   // sa3.0
        {55, 50, 52, 51, 53, 54, 256, 256, 256, 256},   // sa3.1
        {61, 56, 58, 57, 59, 60, 512, 256, 512, 256},   // sa3.2
        { 1,  0, -1, -1, -1, -1,  64, 512,  64, 512},   // conv6
        { 3,  2, -1, -1, -1, -1, 256,  64, 256,  64},   // fc1
        { 5,  4, -1, -1, -1, -1, 512, 256, 512, 256},   // fc2
        { 7,  6, -1, -1, -1, -1,  60, 512,  64, 512},   // fc3 (Cout padded to 64)
    };
    half_t* wp[13]; float* bp[13];
    size_t woff = 0, boff = 0;
    for (int i = 0; i < 13; ++i) {
        wp[i] = wreg + woff; bp[i] = breg + boff;
        woff += (size_t)lws[i].cpad * lws[i].kpad;
        boff += (size_t)lws[i].cpad;
        int total = lws[i].cpad * lws[i].kpad;
        prep_weights_kernel<<<(total + 255) / 256, 256, 0, stream>>>(
            (const float*)d_in[lws[i].wi], (const float*)d_in[lws[i].bi],
            lws[i].gi  >= 0 ? (const float*)d_in[lws[i].gi]  : nullptr,
            lws[i].bei >= 0 ? (const float*)d_in[lws[i].bei] : nullptr,
            lws[i].mi  >= 0 ? (const float*)d_in[lws[i].mi]  : nullptr,
            lws[i].vi  >= 0 ? (const float*)d_in[lws[i].vi]  : nullptr,
            wp[i], bp[i], lws[i].cout, lws[i].cin, lws[i].cpad, lws[i].kpad);
    }

    auto gemm = [&](const half_t* A, int li, half_t* outH, float* outF,
                    int M, int Nreal, int relu) {
        int tiles = (M / 32) * (lws[li].cpad / 64);   // 32x64 strip per wave
        int blocks = (tiles * 32 + 255) / 256;
        wmma_gemm_kernel<<<blocks, 256, 0, stream>>>(
            A, wp[li], bp[li], outH, outF, M, lws[li].kpad, lws[li].cpad, Nreal, relu);
    };

    float* out_pc  = (float*)d_out;              // [32,50,20,3] = 96000
    float* out_gf  = (float*)d_out + 96000;      // [32,50,64]   = 102400
    float* out_l3x = (float*)d_out + 198400;     // [32,50,3]    = 4800

    // ---- SA1 ----
    fps_kernel<<<BATCH, 256, 0, stream>>>(part_pcs, N0, S1, l1_xyz);
    ball_query_kernel<<<(BATCH*S1 + 255)/256, 256, 0, stream>>>(
        part_pcs, l1_xyz, N0, S1, K1, 0.2f*0.2f, idx1);
    {
        size_t tot = (size_t)BATCH * S1 * K1 * 32;
        group_kernel<<<(tot + 255)/256, 256, 0, stream>>>(
            part_pcs, l1_xyz, nullptr, idx1, N0, S1, K1, 0, 32, bufA);
    }
    const int M1 = BATCH * S1 * K1;
    gemm(bufA, 0, bufB, nullptr, M1, 64, 1);
    gemm(bufB, 1, bufA, nullptr, M1, 64, 1);
    gemm(bufA, 2, bufB, nullptr, M1, 128, 1);
    maxpool_kernel<<<(BATCH*S1*128 + 255)/256, 256, 0, stream>>>(bufB, S1, K1, 128, 128, l1_pts);

    // ---- SA2 ----
    fps_kernel<<<BATCH, 256, 0, stream>>>(l1_xyz, S1, S2, l2_xyz);
    ball_query_kernel<<<(BATCH*S2 + 255)/256, 256, 0, stream>>>(
        l1_xyz, l2_xyz, S1, S2, K2, 0.4f*0.4f, idx2);
    {
        size_t tot = (size_t)BATCH * S2 * K2 * 160;
        group_kernel<<<(tot + 255)/256, 256, 0, stream>>>(
            l1_xyz, l2_xyz, l1_pts, idx2, S1, S2, K2, 128, 160, bufA);
    }
    const int M2 = BATCH * S2 * K2;
    gemm(bufA, 3, bufB, nullptr, M2, 128, 1);
    gemm(bufB, 4, bufA, nullptr, M2, 128, 1);
    gemm(bufA, 5, bufB, nullptr, M2, 256, 1);
    maxpool_kernel<<<(BATCH*S2*256 + 255)/256, 256, 0, stream>>>(bufB, S2, K2, 256, 256, l2_pts);

    // ---- SA3 ----
    fps_kernel<<<BATCH, 256, 0, stream>>>(l2_xyz, S2, S3, out_l3x);
    ball_query_kernel<<<(BATCH*S3 + 255)/256, 256, 0, stream>>>(
        l2_xyz, out_l3x, S2, S3, K3, 0.8f*0.8f, idx3);
    {
        size_t tot = (size_t)BATCH * S3 * K3 * 288;
        group_kernel<<<(tot + 255)/256, 256, 0, stream>>>(
            l2_xyz, out_l3x, l2_pts, idx3, S2, S3, K3, 256, 288, bufA);
    }
    const int M3 = BATCH * S3 * K3;   // 102400
    gemm(bufA, 6, bufB, nullptr, M3, 256, 1);
    gemm(bufB, 7, bufA, nullptr, M3, 256, 1);
    gemm(bufA, 8, bufB, nullptr, M3, 512, 1);
    maxpool_kernel<<<(BATCH*S3*512 + 255)/256, 256, 0, stream>>>(bufB, S3, K3, 512, 512, l3_pts);

    // ---- head: conv6 -> fc1 -> fc2 -> fc3 ----
    const int Mg = BATCH * S3;        // 1600
    gemm(l3_pts, 9,  gf16, out_gf, Mg, 64, 0);   // gf (f32 out) + f16 copy
    gemm(gf16,  10, fc1o, nullptr, Mg, 256, 1);
    gemm(fc1o,  11, fc2o, nullptr, Mg, 512, 1);
    gemm(fc2o,  12, nullptr, out_pc, Mg, 60, 0); // pc_offset, Cout 60 of padded 64
}
